// SpectralConv2d_31739808317453
// MI455X (gfx1250) — compile-verified, matrix operand order checked by reading, not run on hardware
//
#include <hip/hip_runtime.h>
#include <cstdint>
#include <cstddef>

// Problem constants (match reference)
constexpr int Bc = 8, Mc = 128, Nc = 128, Ic = 64, Oc = 64, MODESc = 32, Hc = 256;

typedef __attribute__((ext_vector_type(16))) _Float16 v16h;
typedef __attribute__((ext_vector_type(8)))  float    v8f;
typedef _Float16 half_t;

union FragU { v16h v; uint4 u[2]; };

__device__ __forceinline__ v8f vzero8() {
  v8f r;
#pragma unroll
  for (int j = 0; j < 8; ++j) r[j] = 0.0f;
  return r;
}

// Load one 16x32 f16 WMMA fragment from a row-major (rows x K) matrix.
// Per CDNA5 ISA layout: lane L holds row L%16; K pairs {b..b+7} and {b+16..b+23}
// with b = 8*(L/16). Two 16-byte loads per lane.
__device__ __forceinline__ v16h load_frag(const half_t* base, int ld, int row0, int k0, int lane) {
  const int hh = lane >> 4;
  const int rr = lane & 15;
  const half_t* p = base + (size_t)(row0 + rr) * (size_t)ld + (size_t)(k0 + hh * 8);
  FragU f;
  f.u[0] = *reinterpret_cast<const uint4*>(p);
  f.u[1] = *reinterpret_cast<const uint4*>(p + 16);
  return f.v;
}

__device__ __forceinline__ v8f wmma16(v16h a, v16h b, v8f c) {
  return __builtin_amdgcn_wmma_f32_16x16x32_f16(false, a, false, b, (short)0, c, false, false);
}

// ---------------------------------------------------------------------------
// K0: h = x + z, write f16 h[b,i,m,n] and hT[b,i,n,m]
// ---------------------------------------------------------------------------
__global__ void __launch_bounds__(256) prep_kernel(const float* __restrict__ z,
                                                   const float* __restrict__ x,
                                                   half_t* __restrict__ h,
                                                   half_t* __restrict__ ht) {
  int t = blockIdx.x * 256 + threadIdx.x;            // over B*M*N*I = 8388608
  int i = t & 63;
  int n = (t >> 6) & 127;
  int m = (t >> 13) & 127;
  int b = t >> 20;
  float v = z[t] + x[t];
  half_t hv = (half_t)v;
  size_t bi = (size_t)(b * Ic + i);
  h [(bi * Mc + m) * Nc + n] = hv;
  ht[(bi * Nc + n) * Mc + m] = hv;
}

// ---------------------------------------------------------------------------
// K1: build DFT matrices (ortho norm), pre-transposed for B-fragment loads.
// Fyt (64 x 128):  row c<32 -> cos(2pi c n/128)/sqrt(128);  c>=32 -> -sin(...)
// Gt  (128 x 64):  col k<32 -> c_k cos/sqrt(128); k>=32 -> -c_k sin/sqrt(128)
// ---------------------------------------------------------------------------
__global__ void __launch_bounds__(256) dft_kernel(half_t* __restrict__ Fyt,
                                                  half_t* __restrict__ Gt) {
  int t = blockIdx.x * 256 + threadIdx.x;            // 16384
  const float inv = 0.08838834764831845f;            // 1/sqrt(128)
  const float w = 6.283185307179586f / 128.0f;
  if (t < 64 * 128) {
    int c = t >> 7, n = t & 127;
    int k = (c < 32) ? c : c - 32;
    float th = w * (float)(k * n);
    float v = (c < 32) ? __cosf(th) : -__sinf(th);
    Fyt[t] = (half_t)(v * inv);
  } else {
    int t2 = t - 8192;
    int n = t2 >> 6, c = t2 & 63;
    int k = (c < 32) ? c : c - 32;
    float ck = (k == 0) ? 1.0f : 2.0f;
    float th = w * (float)(k * n);
    float v = (c < 32) ? __cosf(th) : -__sinf(th);
    Gt[t2] = (half_t)(v * ck * inv);
  }
}

// ---------------------------------------------------------------------------
// K2: pack complex spectral weights into real 128x128 per-mode matrices,
// stored transposed (cols x K):  Wt[k][c][kk], c<64 Re-out cols, c>=64 Im-out.
//   Re = fr*Wr - fi*Wi ; Im = fr*Wi + fi*Wr
// ---------------------------------------------------------------------------
__global__ void __launch_bounds__(256) pack_spec_kernel(const float* __restrict__ fw,
                                                        half_t* __restrict__ Wt) {
  int t = blockIdx.x * 256 + threadIdx.x;            // 32*128*128 = 524288
  int kk = t & 127;
  int c  = (t >> 7) & 127;
  int k  = t >> 14;
  int o = (c < 64) ? c : c - 64;
  int i = (kk < 64) ? kk : kk - 64;
  const float* base = fw + (((size_t)i * Oc + o) * MODESc + k) * 2;
  float wr = base[0], wi = base[1];
  float v;
  if (c < 64) v = (kk < 64) ? wr : -wi;
  else        v = (kk < 64) ? wi :  wr;
  Wt[t] = (half_t)v;
}

// K2b: transpose FFN weights to (cols x K) f16
__global__ void __launch_bounds__(256) pack_ffn_kernel(const float* __restrict__ w1,
                                                       const float* __restrict__ w2,
                                                       half_t* __restrict__ w1t,
                                                       half_t* __restrict__ w2t) {
  int t = blockIdx.x * 256 + threadIdx.x;            // 32768
  if (t < Hc * Ic) {                                  // w1t[h][i] = w1[i][h]
    int i = t & 63, hcol = t >> 6;
    w1t[t] = (half_t)w1[i * Hc + hcol];
  } else {                                            // w2t[o][h] = w2[h][o]
    int t2 = t - Hc * Ic;
    int hcol = t2 & 255, o = t2 >> 8;
    w2t[t2] = (half_t)w2[hcol * Ic + o];
  }
}

// ---------------------------------------------------------------------------
// K3: forward truncated rfft as GEMM: (65536 x 128) x (128 x 64 cols)
// A rows r = (b*64+i)*128 + d  (d = m or n). Scatter to per-mode layout:
//   S1[((k*8+b)*128 + d)*128 + ph*64 + i]   (ph: 0=Re, 1=Im)
// ---------------------------------------------------------------------------
__global__ void __launch_bounds__(256) fft_fwd_kernel(const half_t* __restrict__ A,
                                                      const half_t* __restrict__ Ft,
                                                      half_t* __restrict__ S1) {
  int wid  = (blockIdx.x * 256 + threadIdx.x) >> 5;  // 0..4095
  int lane = threadIdx.x & 31;
  int row0 = wid * 16;
  v8f acc[4];
#pragma unroll
  for (int c = 0; c < 4; ++c) acc[c] = vzero8();
#pragma unroll
  for (int kk = 0; kk < 128; kk += 32) {
    v16h a = load_frag(A, 128, row0, kk, lane);
#pragma unroll
    for (int c = 0; c < 4; ++c) {
      v16h bf = load_frag(Ft, 128, c * 16, kk, lane);
      acc[c] = wmma16(a, bf, acc[c]);
    }
  }
  int hh = lane >> 4, ln = lane & 15;
#pragma unroll
  for (int c = 0; c < 4; ++c) {
#pragma unroll
    for (int r = 0; r < 8; ++r) {
      int row = row0 + r + 8 * hh;
      int d  = row & 127;
      int bi = row >> 7;
      int i  = bi & 63;
      int b  = bi >> 6;
      int col = c * 16 + ln;
      int k  = (col < 32) ? col : col - 32;
      int ph = (col < 32) ? 0 : 1;
      S1[(((size_t)(k * Bc + b)) * 128 + d) * 128 + ph * 64 + i] = (half_t)acc[c][r];
    }
  }
}

// ---------------------------------------------------------------------------
// K4: per-mode complex mixing: for each k, (1024 x 128) x (128 x 128 cols).
// Scatter to irfft-ready layout: A3[((b*64+o)*128 + d)*64 + ph*32 + k]
// ---------------------------------------------------------------------------
__global__ void __launch_bounds__(256) mode_mix_kernel(const half_t* __restrict__ S1,
                                                       const half_t* __restrict__ Wt,
                                                       half_t* __restrict__ A3) {
  int wid  = (blockIdx.x * 256 + threadIdx.x) >> 5;  // 0..2047
  int lane = threadIdx.x & 31;
  int k  = wid >> 6;                                  // mode
  int tr = wid & 63;                                  // row tile
  const half_t* Ak = S1 + (size_t)k * 1024 * 128;
  const half_t* Bk = Wt + (size_t)k * 128 * 128;
  int row0 = tr * 16;
  v8f acc[8];
#pragma unroll
  for (int c = 0; c < 8; ++c) acc[c] = vzero8();
#pragma unroll
  for (int kk = 0; kk < 128; kk += 32) {
    v16h a = load_frag(Ak, 128, row0, kk, lane);
#pragma unroll
    for (int c = 0; c < 8; ++c) {
      v16h bf = load_frag(Bk, 128, c * 16, kk, lane);
      acc[c] = wmma16(a, bf, acc[c]);
    }
  }
  int hh = lane >> 4, ln = lane & 15;
#pragma unroll
  for (int c = 0; c < 8; ++c) {
#pragma unroll
    for (int r = 0; r < 8; ++r) {
      int row = row0 + r + 8 * hh;                    // b*128 + d
      int d = row & 127;
      int b = row >> 7;
      int col = c * 16 + ln;
      int o  = (col < 64) ? col : col - 64;
      int ph = (col < 64) ? 0 : 1;
      A3[(((size_t)(b * Oc + o)) * 128 + d) * 64 + ph * 32 + k] = (half_t)acc[c][r];
    }
  }
}

// ---------------------------------------------------------------------------
// K5: irfft as GEMM: (65536 x 64) x (64 x 128 cols) -> f16 (b,o,d1,d2)
// ---------------------------------------------------------------------------
__global__ void __launch_bounds__(256) ifft_kernel(const half_t* __restrict__ A3,
                                                   const half_t* __restrict__ Gt,
                                                   half_t* __restrict__ outx) {
  int wid  = (blockIdx.x * 256 + threadIdx.x) >> 5;  // 0..4095
  int lane = threadIdx.x & 31;
  int row0 = wid * 16;
  v8f acc[8];
#pragma unroll
  for (int c = 0; c < 8; ++c) acc[c] = vzero8();
#pragma unroll
  for (int kk = 0; kk < 64; kk += 32) {
    v16h a = load_frag(A3, 64, row0, kk, lane);
#pragma unroll
    for (int c = 0; c < 8; ++c) {
      v16h bf = load_frag(Gt, 64, c * 16, kk, lane);
      acc[c] = wmma16(a, bf, acc[c]);
    }
  }
  int hh = lane >> 4, ln = lane & 15;
#pragma unroll
  for (int c = 0; c < 8; ++c) {
#pragma unroll
    for (int r = 0; r < 8; ++r) {
      int row = row0 + r + 8 * hh;
      outx[(size_t)row * 128 + c * 16 + ln] = (half_t)acc[c][r];
    }
  }
}

// ---------------------------------------------------------------------------
// K6: fused s = xy + xx^T gather, FFN (Linear->ReLU->Linear), bias, +z residual
// One wave per 16-row (b,m,n) tile; LDS stages s (16x64) and hidden (16x256).
// ---------------------------------------------------------------------------
__global__ void __launch_bounds__(128) ffn_kernel(const half_t* __restrict__ xy,
                                                  const half_t* __restrict__ xxT,
                                                  const half_t* __restrict__ w1t,
                                                  const half_t* __restrict__ w2t,
                                                  const float* __restrict__ b1g,
                                                  const float* __restrict__ b2g,
                                                  const float* __restrict__ zg,
                                                  float* __restrict__ out) {
  __shared__ __align__(16) half_t sS[4][16 * 64];
  __shared__ __align__(16) half_t sH[4][16 * 256];
  int w    = threadIdx.x >> 5;
  int lane = threadIdx.x & 31;
  int wid  = blockIdx.x * 4 + w;                      // 0..8191
  int r0   = wid * 16;                                // flat (b,m,n) row base
  int n0   = r0 & 127;
  int bm   = r0 >> 7;
  int m    = bm & 127;
  int b    = bm >> 7;

  // Gather s[row][o] = xy[b,o,m,n0+row] + xxT[b,o,n0+row,m]
#pragma unroll 4
  for (int j = 0; j < 32; ++j) {
    int idx = j * 32 + lane;                          // 0..1023
    int row = idx >> 6, col = idx & 63;
    float vy = (float)xy [(((size_t)(b * Oc + col)) * Mc + m) * Nc + n0 + row];
    float vx = (float)xxT[(((size_t)(b * Oc + col)) * Nc + n0 + row) * Mc + m];
    sS[w][row * 64 + col] = (half_t)(vy + vx);
  }
  __syncthreads();

  int hh = lane >> 4, ln = lane & 15;
  // GEMM1: (16x64) x (64x256) + b1, ReLU -> sH  (4 chunks of 64 cols)
  v16h aS0 = load_frag(sS[w], 64, 0, 0, lane);
  v16h aS1 = load_frag(sS[w], 64, 0, 32, lane);
#pragma unroll
  for (int ch = 0; ch < 4; ++ch) {
    v8f acc[4];
#pragma unroll
    for (int c = 0; c < 4; ++c) acc[c] = vzero8();
#pragma unroll
    for (int c = 0; c < 4; ++c) {
      int col0 = ch * 64 + c * 16;
      v16h bf0 = load_frag(w1t, 64, col0, 0, lane);
      acc[c] = wmma16(aS0, bf0, acc[c]);
      v16h bf1 = load_frag(w1t, 64, col0, 32, lane);
      acc[c] = wmma16(aS1, bf1, acc[c]);
    }
#pragma unroll
    for (int c = 0; c < 4; ++c) {
#pragma unroll
      for (int r = 0; r < 8; ++r) {
        int row = r + 8 * hh;
        int col = ch * 64 + c * 16 + ln;
        float v = acc[c][r] + b1g[col];
        sH[w][row * 256 + col] = (half_t)(v > 0.0f ? v : 0.0f);
      }
    }
  }
  __syncthreads();

  // GEMM2: (16x256) x (256x64) + b2 + z -> out (fp32, layout (b,m,n,i))
  v8f acc2[4];
#pragma unroll
  for (int c = 0; c < 4; ++c) acc2[c] = vzero8();
#pragma unroll
  for (int kk = 0; kk < 256; kk += 32) {
    v16h a = load_frag(sH[w], 256, 0, kk, lane);
#pragma unroll
    for (int c = 0; c < 4; ++c) {
      v16h bf = load_frag(w2t, 256, c * 16, kk, lane);
      acc2[c] = wmma16(a, bf, acc2[c]);
    }
  }
#pragma unroll
  for (int c = 0; c < 4; ++c) {
#pragma unroll
    for (int r = 0; r < 8; ++r) {
      int row = r + 8 * hh;
      size_t dst = (size_t)(r0 + row) * 64 + c * 16 + ln;
      out[dst] = zg[dst] + acc2[c][r] + b2g[c * 16 + ln];
    }
  }
}

// ---------------------------------------------------------------------------
extern "C" void kernel_launch(void* const* d_in, const int* in_sizes, int n_in,
                              void* d_out, int out_size, void* d_ws, size_t ws_size,
                              hipStream_t stream) {
  const float* z   = (const float*)d_in[0];
  const float* x   = (const float*)d_in[1];
  const float* fw1 = (const float*)d_in[2];
  const float* fw2 = (const float*)d_in[3];
  const float* w1  = (const float*)d_in[4];
  const float* b1  = (const float*)d_in[5];
  const float* w2  = (const float*)d_in[6];
  const float* b2  = (const float*)d_in[7];
  float* out = (float*)d_out;
  char* ws = (char*)d_ws;

  // Workspace layout (bytes); A3 buffers reuse the dead h / hT regions.
  const size_t OFF_H   = 0;          // 16 MB  h   (later A3Y)
  const size_t OFF_HT  = 16777216;   // 16 MB  hT  (later A3X)
  const size_t OFF_S1Y = 33554432;   //  8 MB
  const size_t OFF_S1X = 41943040;   //  8 MB
  const size_t OFF_XY  = 50331648;   // 16 MB
  const size_t OFF_XX  = 67108864;   // 16 MB
  const size_t OFF_FY  = 83886080;   // 16 KB
  const size_t OFF_G   = 83902464;   // 16 KB
  const size_t OFF_W1T = 83918848;   //  1 MB
  const size_t OFF_W2T = 84967424;   //  1 MB
  const size_t OFF_W1F = 86016000;   // 32 KB
  const size_t OFF_W2F = 86048768;   // 32 KB  (end: 86081536)

  half_t* H   = (half_t*)(ws + OFF_H);
  half_t* HT  = (half_t*)(ws + OFF_HT);
  half_t* S1Y = (half_t*)(ws + OFF_S1Y);
  half_t* S1X = (half_t*)(ws + OFF_S1X);
  half_t* XY  = (half_t*)(ws + OFF_XY);
  half_t* XX  = (half_t*)(ws + OFF_XX);
  half_t* FYT = (half_t*)(ws + OFF_FY);
  half_t* GT  = (half_t*)(ws + OFF_G);
  half_t* W1T = (half_t*)(ws + OFF_W1T);
  half_t* W2T = (half_t*)(ws + OFF_W2T);
  half_t* W1F = (half_t*)(ws + OFF_W1F);
  half_t* W2F = (half_t*)(ws + OFF_W2F);
  half_t* A3Y = H;    // reuse (h dead after its forward FFT)
  half_t* A3X = HT;   // reuse

  prep_kernel    <<<32768, 256, 0, stream>>>(z, x, H, HT);
  dft_kernel     <<<64,    256, 0, stream>>>(FYT, GT);
  pack_spec_kernel<<<2048, 256, 0, stream>>>(fw1, W1T);
  pack_spec_kernel<<<2048, 256, 0, stream>>>(fw2, W2T);
  pack_ffn_kernel<<<128,   256, 0, stream>>>(w1, w2, W1F, W2F);

  fft_fwd_kernel <<<512, 256, 0, stream>>>(H,  FYT, S1Y);   // along n
  fft_fwd_kernel <<<512, 256, 0, stream>>>(HT, FYT, S1X);   // along m
  mode_mix_kernel<<<256, 256, 0, stream>>>(S1Y, W1T, A3Y);
  mode_mix_kernel<<<256, 256, 0, stream>>>(S1X, W2T, A3X);
  ifft_kernel    <<<512, 256, 0, stream>>>(A3Y, GT, XY);    // xy[b,o,m,n]
  ifft_kernel    <<<512, 256, 0, stream>>>(A3X, GT, XX);    // xx[b,o,n,m]

  ffn_kernel     <<<2048, 128, 0, stream>>>(XY, XX, W1F, W2F, b1, b2, z, out);
}
